// FUDFT_74981539053860
// MI455X (gfx1250) — matrix operand order repaired and checked by reading, not compiled, not run on hardware
//
#include <hip/hip_runtime.h>

typedef __attribute__((ext_vector_type(16))) _Float16 v16h;
typedef __attribute__((ext_vector_type(8)))  float    v8f;

#define Bsz 4
#define Nn  4096
#define Cc  32
#define Ss  4096

// Pass 1: x[b][n][c] (f32) -> xt[b][c][n] (f16), so B-fragments are contiguous in n.
__global__ __launch_bounds__(256) void fudft_transpose_cvt(const float* __restrict__ x,
                                                           _Float16* __restrict__ xt) {
    int idx = blockIdx.x * blockDim.x + threadIdx.x;
    if (idx >= Bsz * Nn * Cc) return;
    int c = idx % Cc;
    int n = (idx / Cc) % Nn;
    int b = idx / (Nn * Cc);
    xt[((size_t)b * Cc + c) * (size_t)Nn + n] = (_Float16)x[idx];
}

// Pass 2: one wave per (batch, 16-row s-tile). Accumulate cos/sin GEMMs with WMMA.
__global__ __launch_bounds__(128) void fudft_wmma(const float* __restrict__ t,
                                                  const float* __restrict__ freqs,
                                                  const _Float16* __restrict__ xt,
                                                  float* __restrict__ out) {
    const int lane = threadIdx.x & 31;
    const int wave = threadIdx.x >> 5;
    const int w    = blockIdx.x * 4 + wave;       // 1024 waves total
    const int b    = w / (Ss / 16);
    const int s0   = (w % (Ss / 16)) * 16;
    const int half = lane >> 4;                   // which half-wave
    const int m    = lane & 15;                   // A row / B column index

    // phase (in turns): freqs[s]*(N-1)/N * t*(N-1). Sign dropped: |z| is invariant.
    const float K1   = ((float)(Nn - 1) * (float)(Nn - 1)) / (float)Nn;
    const float rowp = freqs[s0 + m] * K1;        // turns per unit t

    const float*    tb = t  + (size_t)b * Nn;
    const _Float16* xb = xt + (size_t)b * Cc * Nn;

    v8f acc_re0 = {}, acc_re1 = {}, acc_im0 = {}, acc_im1 = {};

    for (int n0 = 0; n0 < Nn; n0 += 32) {
        // This lane's 16 A-fragment K slots (contiguous, matching B):
        //   half==0 -> K = n0 + {0..15};  half==1 -> K = n0 + {16..31}
        const float4 t0 = *(const float4*)(tb + n0 + half * 16);
        const float4 t1 = *(const float4*)(tb + n0 + half * 16 + 4);
        const float4 t2 = *(const float4*)(tb + n0 + half * 16 + 8);
        const float4 t3 = *(const float4*)(tb + n0 + half * 16 + 12);
        float tc[16];
        tc[0]  = t0.x; tc[1]  = t0.y; tc[2]  = t0.z; tc[3]  = t0.w;
        tc[4]  = t1.x; tc[5]  = t1.y; tc[6]  = t1.z; tc[7]  = t1.w;
        tc[8]  = t2.x; tc[9]  = t2.y; tc[10] = t2.z; tc[11] = t2.w;
        tc[12] = t3.x; tc[13] = t3.y; tc[14] = t3.z; tc[15] = t3.w;

        v16h a_cos, a_sin;
#pragma unroll
        for (int j = 0; j < 16; ++j) {
            float ph = rowp * tc[j];
            ph = ph - __builtin_floorf(ph);                    // fract -> [0,1) turns
            a_cos[j] = (_Float16)__builtin_amdgcn_cosf(ph);    // v_cos_f32 (period 1)
            a_sin[j] = (_Float16)__builtin_amdgcn_sinf(ph);    // v_sin_f32
        }

        // B fragments (contiguous k order, same as A): lane m = column (channel),
        // lanes 0-15 hold K = n0..n0+15, lanes 16-31 hold K = n0+16..n0+31.
        const int kb = n0 + half * 16;
        v16h bf0 = *(const v16h*)(xb + (size_t)m        * Nn + kb);  // channels 0..15
        v16h bf1 = *(const v16h*)(xb + (size_t)(m + 16) * Nn + kb);  // channels 16..31

        acc_re0 = __builtin_amdgcn_wmma_f32_16x16x32_f16(false, a_cos, false, bf0,
                                                         (short)0, acc_re0, false, false);
        acc_re1 = __builtin_amdgcn_wmma_f32_16x16x32_f16(false, a_cos, false, bf1,
                                                         (short)0, acc_re1, false, false);
        acc_im0 = __builtin_amdgcn_wmma_f32_16x16x32_f16(false, a_sin, false, bf0,
                                                         (short)0, acc_im0, false, false);
        acc_im1 = __builtin_amdgcn_wmma_f32_16x16x32_f16(false, a_sin, false, bf1,
                                                         (short)0, acc_im1, false, false);
    }

    // Epilogue: f32 C/D layout -> lane m is column (channel), VGPR r is row r + 8*half.
    const float inv = 0.015625f;  // 1/sqrt(4096)
#pragma unroll
    for (int r = 0; r < 8; ++r) {
        const int s = s0 + r + half * 8;
        // Branchless zeroing of rows 0 and S-1 (avoid exec-mask divergence).
        const float keep = (s == 0 || s == Ss - 1) ? 0.0f : inv;
        float v0 = __builtin_amdgcn_sqrtf(acc_re0[r] * acc_re0[r] + acc_im0[r] * acc_im0[r]) * keep;
        float v1 = __builtin_amdgcn_sqrtf(acc_re1[r] * acc_re1[r] + acc_im1[r] * acc_im1[r]) * keep;
        out[((size_t)b * Ss + s) * Cc + m]      = v0;
        out[((size_t)b * Ss + s) * Cc + m + 16] = v1;
    }
}

extern "C" void kernel_launch(void* const* d_in, const int* in_sizes, int n_in,
                              void* d_out, int out_size, void* d_ws, size_t ws_size,
                              hipStream_t stream) {
    (void)in_sizes; (void)n_in; (void)out_size; (void)ws_size;
    const float* x     = (const float*)d_in[0];   // [B,N,C] f32
    const float* t     = (const float*)d_in[1];   // [B,N]   f32
    const float* freqs = (const float*)d_in[2];   // [S]     f32
    float*       out   = (float*)d_out;           // [B,S,C] f32
    _Float16*    xt    = (_Float16*)d_ws;         // [B,C,N] f16 (1 MB)

    const int total = Bsz * Nn * Cc;
    fudft_transpose_cvt<<<(total + 255) / 256, 256, 0, stream>>>(x, xt);

    const int nwaves = Bsz * (Ss / 16);           // 1024 waves, 4 per block
    fudft_wmma<<<nwaves / 4, 128, 0, stream>>>(t, freqs, xt, out);
}
